// Head_56057913147637
// MI455X (gfx1250) — matrix operand order repaired and checked by reading, not compiled, not run on hardware
//
#include <hip/hip_runtime.h>
#include <hip/hip_bf16.h>

// Attention head for MI455X (gfx1250, wave32, WMMA 16x16x32 f16).
// One workgroup per batch element b (2048 blocks x 256 threads = 8 waves).
//
// Prep kernel pre-swizzles Wq/Wk/Wv (f32) into f16 WMMA B-fragments in d_ws:
//   ws[((proj*12 + ks)*4 + nt)*32 + lane][0..15] = W[(ks*32 + 16*(lane/16) + h)*64
//                                                    + nt*16 + lane%16]
// so the GEMM inner loop loads each B fragment as two contiguous b128s.
//
// LDS layout (dynamic, 73728 bytes):
//   [     0 .. 49152)  Xs : f16 x[b] tile, 64x384      (dead after phase 2)
//     alias [    0..16384)  Ss : f32 scores 64x64      (phase 3/4)
//     alias [16384..24576)  Ps : f16 probs  64x64      (phase 4/5)
//   [ 49152 .. 57344)  Qs : f16 64x64  [t][h]
//   [ 57344 .. 65536)  Ks : f16 64x64  [t][h]
//   [ 65536 .. 73728)  Vt : f16 64x64  [h][t]  (transposed for phase-5 B frags)

typedef __attribute__((ext_vector_type(16))) _Float16 v16h;
typedef __attribute__((ext_vector_type(8)))  _Float16 v8h;
typedef __attribute__((ext_vector_type(8)))  float    v8f;

#define ATT_B 2048
#define ATT_T 64
#define ATT_C 384
#define ATT_H 64
#define LDS_BYTES 73728
#define NFRAG (3 * 12 * 4 * 32)      // 4608 fragment-lanes, 16 halves each

union AFrag { v16h v; v8h h[2]; };

__device__ __forceinline__ v8f wmma_f16(v16h a, v16h b, v8f c) {
    return __builtin_amdgcn_wmma_f32_16x16x32_f16(
        /*neg_a=*/false, a, /*neg_b=*/false, b,
        /*c_mod=*/(short)0, c, /*reuse_a=*/false, /*reuse_b=*/false);
}

// ---- one-shot weight pre-swizzle: f32 (C,H) -> f16 per-lane B fragments ----
__global__ void __launch_bounds__(256)
pack_weights_kernel(const float* __restrict__ Wq,
                    const float* __restrict__ Wk,
                    const float* __restrict__ Wv,
                    _Float16* __restrict__ wpk) {
    const int idx = blockIdx.x * 256 + threadIdx.x;     // 0..4607
    if (idx >= NFRAG) return;
    const int lane = idx & 31;
    const int nt   = (idx >> 5) & 3;
    const int pk12 = idx >> 7;                          // proj*12 + ks
    const int ks   = pk12 % 12;
    const int proj = idx / (12 * 4 * 32);
    const float* W = (proj == 0) ? Wq : ((proj == 1) ? Wk : Wv);
    const int n  = nt * 16 + (lane & 15);
    const int kb = ks * 32 + ((lane >> 4) << 4);
    _Float16* d = wpk + (size_t)idx * 16;
    #pragma unroll
    for (int h = 0; h < 16; ++h)
        d[h] = (_Float16)W[(kb + h) * ATT_H + n];
}

__global__ void __launch_bounds__(256)
attn_head_kernel(const float* __restrict__ x,
                 const _Float16* __restrict__ wpk,
                 float* __restrict__ out) {
    extern __shared__ char smem[];
    _Float16* Xs  = (_Float16*)smem;                    // 64 x 384
    float*    Ss  = (float*)smem;                       // 64 x 64 (aliases Xs)
    _Float16* Ps  = (_Float16*)(smem + 16384);          // 64 x 64 (aliases Xs)
    _Float16* Qs  = (_Float16*)(smem + 49152);          // 64 x 64 [t][h]
    _Float16* Ksm = (_Float16*)(smem + 57344);          // 64 x 64 [t][h]
    _Float16* Vt  = (_Float16*)(smem + 65536);          // 64 x 64 [h][t]

    const int tid  = threadIdx.x;
    const int lane = tid & 31;
    const int wave = tid >> 5;
    const int b    = blockIdx.x;

    const int lmod = lane & 15;           // row/col within 16
    const int lhi  = lane >> 4;           // 0 or 1 (lane half)
    const int hiA  = lhi << 4;            // A-frag K offset: 0 or 16
    const int hiB  = lhi << 4;            // B-frag K offset: 0 or 16
    const int mhi  = lhi << 3;            // C/D M offset: 0 or 8

    // ---------------- Phase 1: stage x[b] into LDS as f16 -----------------
    {
        const float4* xv = (const float4*)(x + (size_t)b * (ATT_T * ATT_C));
        for (int i = tid; i < (ATT_T * ATT_C) / 4; i += 256) {
            float4 f = xv[i];
            _Float16* d = &Xs[i * 4];
            d[0] = (_Float16)f.x; d[1] = (_Float16)f.y;
            d[2] = (_Float16)f.z; d[3] = (_Float16)f.w;
        }
    }
    __syncthreads();

    // ---------------- Phase 2: Q/K/V = x @ W (48 tiles, 6 per wave) -------
    for (int i = 0; i < 6; ++i) {
        const int t48  = wave * 6 + i;          // 0..47
        const int proj = t48 >> 4;              // 0=Q 1=K 2=V
        const int mt   = (t48 & 15) >> 2;       // M tile 0..3
        const int nt   = t48 & 3;               // N tile 0..3

        const int arow = (mt * 16 + lmod) * ATT_C;   // A: M = lane%16
        const int ncol = nt * 16 + lmod;             // B: N = lane%16
        v8f acc = {};
        for (int ks = 0; ks < ATT_C / 32; ++ks) {
            AFrag a, bm;
            const _Float16* ap = &Xs[arow + ks * 32 + hiA];
            a.h[0] = *(const v8h*)(ap);
            a.h[1] = *(const v8h*)(ap + 8);
            // pre-swizzled B fragment: 32 contiguous bytes per lane
            const _Float16* bp =
                wpk + ((size_t)(((proj * 12 + ks) * 4 + nt) * 32 + lane) * 16);
            bm.h[0] = *(const v8h*)(bp);
            bm.h[1] = *(const v8h*)(bp + 8);
            acc = wmma_f16(a.v, bm.v, acc);
        }
        const int mbase = mt * 16 + mhi;
        if (proj == 2) {
            // V stored transposed [h][t]: 8 accumulator rows are contiguous
            v8h pk;
            #pragma unroll
            for (int r = 0; r < 8; ++r) pk[r] = (_Float16)acc[r];
            *(v8h*)&Vt[ncol * ATT_T + mbase] = pk;
        } else {
            _Float16* dst = (proj == 0) ? Qs : Ksm;
            #pragma unroll
            for (int r = 0; r < 8; ++r)
                dst[(mbase + r) * ATT_H + ncol] = (_Float16)acc[r];
        }
    }
    __syncthreads();

    // ---------------- Phase 3: S = Q @ K^T * C^-0.5, causal mask ----------
    const float scale = 0.05103103630798287f;   // 384^-0.5
    for (int i = 0; i < 2; ++i) {
        const int t16 = wave * 2 + i;           // 0..15
        const int mt = t16 >> 2, nt = t16 & 3;
        const int arow = (mt * 16 + lmod) * ATT_H;
        const int brow = (nt * 16 + lmod) * ATT_H;   // B(K,N) = K[N][K]: contiguous
        v8f acc = {};
        #pragma unroll
        for (int ks = 0; ks < 2; ++ks) {
            AFrag a, bm;
            const _Float16* ap = &Qs[arow + ks * 32 + hiA];
            a.h[0] = *(const v8h*)(ap);
            a.h[1] = *(const v8h*)(ap + 8);
            const _Float16* bp = &Ksm[brow + ks * 32 + hiB];
            bm.h[0] = *(const v8h*)(bp);
            bm.h[1] = *(const v8h*)(bp + 8);
            acc = wmma_f16(a.v, bm.v, acc);
        }
        const int mbase = mt * 16 + mhi;
        const int N = nt * 16 + lmod;
        #pragma unroll
        for (int r = 0; r < 8; ++r) {
            const int t = mbase + r;
            Ss[t * ATT_T + N] = (N <= t) ? acc[r] * scale : -INFINITY;
        }
    }
    __syncthreads();

    // ---------------- Phase 4: row softmax (64 rows, 1 thread each) -------
    if (tid < ATT_T) {
        const int t = tid;
        float mx = -INFINITY;
        for (int s = 0; s <= t; ++s) mx = fmaxf(mx, Ss[t * ATT_T + s]);
        float sum = 0.f;
        for (int s = 0; s <= t; ++s) {
            float e = __expf(Ss[t * ATT_T + s] - mx);
            Ss[t * ATT_T + s] = e;
            sum += e;
        }
        const float inv = 1.f / sum;
        for (int s = 0; s < ATT_T; ++s)
            Ps[t * ATT_T + s] = (s <= t) ? (_Float16)(Ss[t * ATT_T + s] * inv)
                                         : (_Float16)0.f;
    }
    __syncthreads();

    // ---------------- Phase 5: O = P @ V, write f32 to global -------------
    float* outb = out + (size_t)b * (ATT_T * ATT_H);
    for (int i = 0; i < 2; ++i) {
        const int t16 = wave * 2 + i;
        const int mt = t16 >> 2, nt = t16 & 3;
        const int arow = (mt * 16 + lmod) * ATT_T;
        const int ncol = nt * 16 + lmod;
        v8f acc = {};
        #pragma unroll
        for (int ks = 0; ks < 2; ++ks) {
            AFrag a, bm;
            const _Float16* ap = &Ps[arow + ks * 32 + hiA];
            a.h[0] = *(const v8h*)(ap);
            a.h[1] = *(const v8h*)(ap + 8);
            // B(K=s, N=h) = V[s][h] = Vt[h][s]: contiguous in transposed store
            const _Float16* bp = &Vt[ncol * ATT_T + ks * 32 + hiB];
            bm.h[0] = *(const v8h*)(bp);
            bm.h[1] = *(const v8h*)(bp + 8);
            acc = wmma_f16(a.v, bm.v, acc);
        }
        const int mbase = mt * 16 + mhi;
        #pragma unroll
        for (int r = 0; r < 8; ++r)
            outb[(mbase + r) * ATT_H + ncol] = acc[r];
    }
}

extern "C" void kernel_launch(void* const* d_in, const int* in_sizes, int n_in,
                              void* d_out, int out_size, void* d_ws, size_t ws_size,
                              hipStream_t stream) {
    const float* x  = (const float*)d_in[0];
    const float* Wq = (const float*)d_in[1];
    const float* Wk = (const float*)d_in[2];
    const float* Wv = (const float*)d_in[3];
    float* out = (float*)d_out;
    _Float16* wpk = (_Float16*)d_ws;    // 4608*16 f16 = 147456 B of scratch

    // Allow >64KB dynamic LDS (WGP has 320KB). Host-side attr set, not a
    // stream op, so it is graph-capture safe and deterministic.
    hipFuncSetAttribute((const void*)attn_head_kernel,
                        hipFuncAttributeMaxDynamicSharedMemorySize, LDS_BYTES);

    pack_weights_kernel<<<(NFRAG + 255) / 256, 256, 0, stream>>>(Wq, Wk, Wv, wpk);
    attn_head_kernel<<<ATT_B, 256, LDS_BYTES, stream>>>(x, wpk, out);
}
